// MTDecoder_69209103008291
// MI455X (gfx1250) — matrix operand order repaired and checked by reading, not compiled, not run on hardware
//
#include <hip/hip_runtime.h>
#include <hip/hip_bf16.h>
#include <math.h>

typedef __attribute__((ext_vector_type(16))) __bf16 v16bf;
typedef __attribute__((ext_vector_type(8)))  __bf16 v8bf;
typedef __attribute__((ext_vector_type(8)))  float  v8f;

#define VOCAB   32000
#define EMBED   512
#define CTXW    512
#define HID     1024
#define BATCH   32
#define MAX_LEN 64
#define START_T 1
#define GATE    (4*HID)   // 4096
#define KCAT    (2*HID)   // 2048

typedef unsigned short ushort_t;

__device__ __forceinline__ ushort_t f2bf(float f) {
    unsigned int u = __float_as_uint(f);
    u += 0x7FFFu + ((u >> 16) & 1u);          // round-to-nearest-even
    return (ushort_t)(u >> 16);
}
__device__ __forceinline__ float sigm(float x) { return 1.0f / (1.0f + __expf(-x)); }

// ---------------------------------------------------------------------------
// One-time (per launch) conversion kernels
// ---------------------------------------------------------------------------
__global__ void convert_cat_kernel(const float* __restrict__ Wih,
                                   const float* __restrict__ Whh,
                                   ushort_t* __restrict__ Wcat) {
    int idx = blockIdx.x * blockDim.x + threadIdx.x;   // GATE*KCAT threads
    int n = idx >> 11;           // /2048
    int k = idx & 2047;
    float v = (k < HID) ? Wih[(size_t)n * HID + k] : Whh[(size_t)n * HID + (k - HID)];
    Wcat[idx] = f2bf(v);
}

__global__ void convert_bf_kernel(const float* __restrict__ src,
                                  ushort_t* __restrict__ dst, int total) {
    int idx = blockIdx.x * blockDim.x + threadIdx.x;
    if (idx < total) dst[idx] = f2bf(src[idx]);
}

__global__ void add_bias_kernel(const float* __restrict__ a,
                                const float* __restrict__ b,
                                float* __restrict__ o, int n) {
    int idx = blockIdx.x * blockDim.x + threadIdx.x;
    if (idx < n) o[idx] = a[idx] + b[idx];
}

// Initialize state: A1 = 0, A0[:,512:1024] = bf16(ctx), A0[:,1024:2048] = 0,
// c0 = c1 = 0, H1bf = 0, x = START
__global__ void init_kernel(ushort_t* __restrict__ A0, ushort_t* __restrict__ A1,
                            ushort_t* __restrict__ H1bf,
                            float* __restrict__ c0, float* __restrict__ c1,
                            int* __restrict__ x, const float* __restrict__ ctx) {
    int idx = blockIdx.x * blockDim.x + threadIdx.x;   // BATCH*KCAT = 65536
    int b = idx >> 11;
    int j = idx & 2047;
    A1[idx] = 0;
    if (j >= CTXW) {
        A0[idx] = (j < 2 * CTXW) ? f2bf(ctx[(size_t)b * CTXW + (j - CTXW)]) : (ushort_t)0;
    }
    if (j < HID) {
        c0[b * HID + j]  = 0.0f;
        c1[b * HID + j]  = 0.0f;
        H1bf[b * HID + j] = 0;
    }
    if (idx < BATCH) x[idx] = START_T;
}

// Per step: gather emb[x[b]] into A0 columns [0,512)
__global__ void embed_kernel(const float* __restrict__ emb, const int* __restrict__ x,
                             ushort_t* __restrict__ A0) {
    int idx = blockIdx.x * blockDim.x + threadIdx.x;   // BATCH*EMBED = 16384
    int b = idx >> 9;
    int j = idx & 511;
    A0[(size_t)b * KCAT + j] = f2bf(emb[(size_t)x[b] * EMBED + j]);
}

// ---------------------------------------------------------------------------
// WMMA GEMM: out[32 x N] = A[32 x K] * W[N x K]^T + bias
// block = 128 threads (4 waves); each wave computes a 32x32 output tile
// (2 M-tiles x 2 N-tiles) = 4 v_wmma_f32_16x16x32_bf16 per K-step.
// A layout (16-bit A 16x32, ISA 7.12.2): lane half h holds K {8h..8h+7} and
// {16+8h..16+8h+7}. B layout: lane r = column, same K order as A
// -> two 16-byte loads.
// ---------------------------------------------------------------------------
template<int K>
__global__ __launch_bounds__(128) void gemm_wmma_kernel(
        const ushort_t* __restrict__ A, const ushort_t* __restrict__ W,
        const float* __restrict__ bias, float* __restrict__ out, int N) {
    const int lane = threadIdx.x & 31;
    const int wave = threadIdx.x >> 5;
    const int h = lane >> 4;
    const int r = lane & 15;
    const int n0 = blockIdx.x * 128 + wave * 32;
    const int nc0 = n0 + r;          // column for N-tile 0
    const int nc1 = n0 + 16 + r;     // column for N-tile 1

    const ushort_t* arow_lo = A + (size_t)r        * K;
    const ushort_t* arow_hi = A + (size_t)(r + 16) * K;
    const ushort_t* w0 = W + (size_t)nc0 * K;
    const ushort_t* w1 = W + (size_t)nc1 * K;

    v8f acc00 = {}, acc10 = {}, acc01 = {}, acc11 = {};

    for (int kb = 0; kb < K; kb += 32) {
        const int o1 = kb + 8 * h;
        const int o2 = kb + 16 + 8 * h;
        v8bf al0 = *(const v8bf*)(arow_lo + o1);
        v8bf al1 = *(const v8bf*)(arow_lo + o2);
        v8bf ah0 = *(const v8bf*)(arow_hi + o1);
        v8bf ah1 = *(const v8bf*)(arow_hi + o2);
        v16bf a_lo = __builtin_shufflevector(al0, al1, 0,1,2,3,4,5,6,7,8,9,10,11,12,13,14,15);
        v16bf a_hi = __builtin_shufflevector(ah0, ah1, 0,1,2,3,4,5,6,7,8,9,10,11,12,13,14,15);
        v8bf b00 = *(const v8bf*)(w0 + o1);
        v8bf b01 = *(const v8bf*)(w0 + o2);
        v8bf b10 = *(const v8bf*)(w1 + o1);
        v8bf b11 = *(const v8bf*)(w1 + o2);
        v16bf b0 = __builtin_shufflevector(b00, b01, 0,1,2,3,4,5,6,7,8,9,10,11,12,13,14,15);
        v16bf b1 = __builtin_shufflevector(b10, b11, 0,1,2,3,4,5,6,7,8,9,10,11,12,13,14,15);

        acc00 = __builtin_amdgcn_wmma_f32_16x16x32_bf16(false, a_lo, false, b0,
                                                        (short)0, acc00, false, false);
        acc10 = __builtin_amdgcn_wmma_f32_16x16x32_bf16(false, a_hi, false, b0,
                                                        (short)0, acc10, false, false);
        acc01 = __builtin_amdgcn_wmma_f32_16x16x32_bf16(false, a_lo, false, b1,
                                                        (short)0, acc01, false, false);
        acc11 = __builtin_amdgcn_wmma_f32_16x16x32_bf16(false, a_hi, false, b1,
                                                        (short)0, acc11, false, false);
    }

    const float bv0 = bias[nc0];
    const float bv1 = bias[nc1];
    // C/D layout: VGPR j, lanes 0-15 -> M=j, lanes 16-31 -> M=j+8
    #pragma unroll
    for (int j = 0; j < 8; ++j) {
        const int m = j + 8 * h;
        out[(size_t)m        * N + nc0] = acc00[j] + bv0;
        out[(size_t)(m + 16) * N + nc0] = acc10[j] + bv0;
        out[(size_t)m        * N + nc1] = acc01[j] + bv1;
        out[(size_t)(m + 16) * N + nc1] = acc11[j] + bv1;
    }
}

// ---------------------------------------------------------------------------
// LSTM pointwise: gates (32x4096 f32) + c -> new c; h written as bf16 into
// two destination matrices (pre-offset column bases, given row strides).
// ---------------------------------------------------------------------------
__global__ void lstm_pointwise_kernel(const float* __restrict__ gates,
                                      float* __restrict__ c,
                                      ushort_t* __restrict__ dst1, int s1,
                                      ushort_t* __restrict__ dst2, int s2) {
    int idx = blockIdx.x * blockDim.x + threadIdx.x;   // BATCH*HID = 32768
    int b = idx >> 10;
    int j = idx & 1023;
    const float* g = gates + (size_t)b * GATE;
    float iv = sigm(g[j]);
    float fv = sigm(g[j + HID]);
    float gv = tanhf(g[j + 2 * HID]);
    float ov = sigm(g[j + 3 * HID]);
    float cn = fv * c[idx] + iv * gv;
    c[idx] = cn;
    ushort_t hb = f2bf(ov * tanhf(cn));
    dst1[(size_t)b * s1 + j] = hb;
    dst2[(size_t)b * s2 + j] = hb;
}

// ---------------------------------------------------------------------------
// Softmax + argmax over VOCAB per batch row; writes probs into d_out[b][t][:]
// and the argmax token into x (first-max tie-break like jnp.argmax).
// ---------------------------------------------------------------------------
__global__ __launch_bounds__(512) void softmax_argmax_kernel(
        const float* __restrict__ logits, float* __restrict__ out, int t,
        int* __restrict__ x_next) {
    const int b = blockIdx.x;
    const int tid = threadIdx.x;
    const float* row = logits + (size_t)b * VOCAB;
    float* prow = out + ((size_t)b * MAX_LEN + t) * VOCAB;

    __shared__ float sval[512];
    __shared__ int   sidx[512];

    float m = -3.4e38f; int mi = VOCAB;
    for (int v = tid; v < VOCAB; v += 512) {
        float val = row[v];
        if (val > m) { m = val; mi = v; }
    }
    sval[tid] = m; sidx[tid] = mi;
    __syncthreads();
    for (int s = 256; s > 0; s >>= 1) {
        if (tid < s) {
            float ov = sval[tid + s]; int oi = sidx[tid + s];
            if (ov > sval[tid] || (ov == sval[tid] && oi < sidx[tid])) {
                sval[tid] = ov; sidx[tid] = oi;
            }
        }
        __syncthreads();
    }
    const float M = sval[0];
    const int   MI = sidx[0];
    __syncthreads();

    float total = 0.0f;
    for (int v = tid; v < VOCAB; v += 512) total += __expf(row[v] - M);
    sval[tid] = total;
    __syncthreads();
    for (int s = 256; s > 0; s >>= 1) {
        if (tid < s) sval[tid] += sval[tid + s];
        __syncthreads();
    }
    const float inv = 1.0f / sval[0];
    for (int v = tid; v < VOCAB; v += 512) prow[v] = __expf(row[v] - M) * inv;
    if (tid == 0) x_next[b] = MI;
}

// ---------------------------------------------------------------------------
extern "C" void kernel_launch(void* const* d_in, const int* in_sizes, int n_in,
                              void* d_out, int out_size, void* d_ws, size_t ws_size,
                              hipStream_t stream) {
    (void)in_sizes; (void)n_in; (void)out_size; (void)ws_size;
    const float* ctx    = (const float*)d_in[0];
    const float* emb    = (const float*)d_in[2];
    const float* W_ih0  = (const float*)d_in[3];
    const float* W_hh0  = (const float*)d_in[4];
    const float* b_ih0  = (const float*)d_in[5];
    const float* b_hh0  = (const float*)d_in[6];
    const float* W_ih1  = (const float*)d_in[7];
    const float* W_hh1  = (const float*)d_in[8];
    const float* b_ih1  = (const float*)d_in[9];
    const float* b_hh1  = (const float*)d_in[10];
    const float* W_lin  = (const float*)d_in[11];
    const float* b_lin  = (const float*)d_in[12];
    float* out = (float*)d_out;

    // workspace layout (256B aligned slices)
    char* p = (char*)d_ws;
    auto carve = [&](size_t bytes) -> char* {
        char* q = p; p += (bytes + 255) & ~(size_t)255; return q;
    };
    ushort_t* W0cat = (ushort_t*)carve((size_t)GATE * KCAT * 2);   // 16.8 MB
    ushort_t* W1cat = (ushort_t*)carve((size_t)GATE * KCAT * 2);   // 16.8 MB
    ushort_t* WL    = (ushort_t*)carve((size_t)VOCAB * HID * 2);   // 65.5 MB
    ushort_t* A0    = (ushort_t*)carve((size_t)BATCH * KCAT * 2);  // [emb|ctx|h0]
    ushort_t* A1    = (ushort_t*)carve((size_t)BATCH * KCAT * 2);  // [h0|h1]
    ushort_t* H1bf  = (ushort_t*)carve((size_t)BATCH * HID * 2);
    float*    gates = (float*)   carve((size_t)BATCH * GATE * 4);
    float*    logit = (float*)   carve((size_t)BATCH * VOCAB * 4); // 4 MB
    float*    c0    = (float*)   carve((size_t)BATCH * HID * 4);
    float*    c1    = (float*)   carve((size_t)BATCH * HID * 4);
    float*    bias0 = (float*)   carve((size_t)GATE * 4);
    float*    bias1 = (float*)   carve((size_t)GATE * 4);
    int*      x     = (int*)     carve((size_t)BATCH * 4);

    // one-time weight conversion (bf16 weights ~99MB -> resident in 192MB L2)
    convert_cat_kernel<<<(GATE * KCAT) / 256, 256, 0, stream>>>(W_ih0, W_hh0, W0cat);
    convert_cat_kernel<<<(GATE * KCAT) / 256, 256, 0, stream>>>(W_ih1, W_hh1, W1cat);
    convert_bf_kernel<<<(VOCAB * HID + 255) / 256, 256, 0, stream>>>(W_lin, WL, VOCAB * HID);
    add_bias_kernel<<<GATE / 256, 256, 0, stream>>>(b_ih0, b_hh0, bias0, GATE);
    add_bias_kernel<<<GATE / 256, 256, 0, stream>>>(b_ih1, b_hh1, bias1, GATE);
    init_kernel<<<(BATCH * KCAT) / 256, 256, 0, stream>>>(A0, A1, H1bf, c0, c1, x, ctx);

    for (int t = 0; t < MAX_LEN; ++t) {
        embed_kernel<<<(BATCH * EMBED) / 256, 256, 0, stream>>>(emb, x, A0);

        // cell 0: gates = [emb|ctx|h0] @ [W_ih0|W_hh0]^T + (b_ih0+b_hh0)
        gemm_wmma_kernel<KCAT><<<GATE / 128, 128, 0, stream>>>(A0, W0cat, bias0, gates, GATE);
        // h0 -> A1 cols[0:1024) (this step, cell1) and A0 cols[1024:2048) (next step)
        lstm_pointwise_kernel<<<(BATCH * HID) / 256, 256, 0, stream>>>(
            gates, c0, A1, KCAT, A0 + HID, KCAT);

        // cell 1: gates = [h0|h1] @ [W_ih1|W_hh1]^T + (b_ih1+b_hh1)
        gemm_wmma_kernel<KCAT><<<GATE / 128, 128, 0, stream>>>(A1, W1cat, bias1, gates, GATE);
        // h1 -> A1 cols[1024:2048) (next step) and compact H1bf (vocab GEMM)
        lstm_pointwise_kernel<<<(BATCH * HID) / 256, 256, 0, stream>>>(
            gates, c1, A1 + HID, KCAT, H1bf, HID);

        // vocab projection: logits = h1 @ W_lin^T + b_lin
        gemm_wmma_kernel<HID><<<VOCAB / 128, 128, 0, stream>>>(H1bf, WL, b_lin, logit, VOCAB);

        softmax_argmax_kernel<<<BATCH, 512, 0, stream>>>(logit, out, t, x);
    }
}
